// BlockSelfAttention_64450279244498
// MI455X (gfx1250) — matrix operand order repaired and checked by reading, not compiled, not run on hardware
//
#include <hip/hip_runtime.h>

// Block self-attention kernel for MI455X (gfx1250, wave32).
// Roofline: ~140 GFLOP vs ~268 MB HBM traffic -> matrix-core bound.
// Both GEMMs (Gram + weighted sum) go through v_wmma_f32_16x16x32_bf16.

#define DFEAT   128
#define BLK     1024
#define ROWS_WG 128      // rows of one block handled per workgroup (8 waves x 16)
#define JT      32       // j-tile width == WMMA K depth for bf16

typedef __attribute__((ext_vector_type(16))) __bf16        v16bf;
typedef __attribute__((ext_vector_type(8)))  float         v8f;
typedef __attribute__((ext_vector_type(8)))  unsigned int  v8u;
typedef __attribute__((ext_vector_type(2)))  __bf16        bf16x2;

static __device__ __forceinline__ unsigned pack2(float x, float y) {
    bf16x2 p; p[0] = (__bf16)x; p[1] = (__bf16)y;
    return __builtin_bit_cast(unsigned, p);
}

__global__ __launch_bounds__(256)
void block_attn_wmma(const float* __restrict__ feat, float* __restrict__ out) {
    // LDS: row strides padded so 16-lane fragment loads hit distinct banks.
    __shared__ __bf16 jtile [JT   ][DFEAT + 4]; // j-tile, row-major (B-frags for S = X*X^T)
    __shared__ __bf16 jtileT[DFEAT][JT    + 4]; // j-tile, transposed (B-frags for O = W*X)
    __shared__ __bf16 wtile [8][16][JT    + 4]; // per-wave W re-layout bounce (D-layout -> A-layout)
    __shared__ float  sqj[BLK];                 // squared norms of all rows of this block

    const int t    = threadIdx.x;
    const int lane = t & 31;
    const int wave = t >> 5;
    const int mrow = lane & 15;     // fragment row/col index within 16
    const int half = lane >> 4;     // 0: lanes 0-15, 1: lanes 16-31
    const int blk  = blockIdx.x >> 3;             // 8 WGs per 1024-row block
    const int wgRow = (blockIdx.x & 7) * ROWS_WG; // this WG's first row in block
    const long long blkRow0 = (long long)blk * BLK;

    // ---- Phase 0: squared norms of all 1024 rows (8 lanes per row, shfl reduce)
    {
        const int seg  = lane & 7;   // 16-float segment within a row
        const int rsub = lane >> 3;  // 4 rows per wave per pass
        for (int pass = 0; pass < BLK / 32; ++pass) {
            int r = pass * 32 + wave * 4 + rsub;
            const float4* p = reinterpret_cast<const float4*>(
                feat + (blkRow0 + r) * DFEAT + seg * 16);
            float s = 0.f;
            #pragma unroll
            for (int q = 0; q < 4; ++q) {
                float4 v = p[q];
                s += v.x * v.x + v.y * v.y + v.z * v.z + v.w * v.w;
            }
            s += __shfl_xor(s, 1);
            s += __shfl_xor(s, 2);
            s += __shfl_xor(s, 4);
            if (seg == 0) sqj[r] = s;
        }
    }
    __syncthreads();

    // ---- Phase 1: wave's 16x128 A-fragments of X (bf16), pinned in registers.
    // 16-bit A layout: VGPR i holds K pair (i>>2)*16 + half*8 + (i&3)*2.
    const int rLoc = wgRow + wave * 16;
    v16bf aX[4];
    #pragma unroll
    for (int f = 0; f < 4; ++f) {
        v8u u;
        const float* rp = feat + (blkRow0 + rLoc + mrow) * DFEAT;
        #pragma unroll
        for (int i = 0; i < 8; ++i) {
            int K = f * 32 + (i >> 2) * 16 + half * 8 + (i & 3) * 2;
            u[i] = pack2(rp[K], rp[K + 1]);
        }
        aX[f] = __builtin_bit_cast(v16bf, u);
    }

    // sq of this wave's rows, indexed per the C/D layout (M = v + 8*half)
    float si[8];
    #pragma unroll
    for (int v = 0; v < 8; ++v) si[v] = sqj[rLoc + v + half * 8];

    v8f accO[8];
    #pragma unroll
    for (int n = 0; n < 8; ++n) { v8f z = {}; accO[n] = z; }

    const float kscale = 1.0f / 25.6f; // 1 / (2 * D/10)

    // ---- Main loop: 32 j-tiles of 32 columns
    for (int jb = 0; jb < BLK; jb += JT) {
        // Stage j-tile into LDS (row-major bf16 + transposed bf16), cooperatively.
        {
            int jr   = t >> 3;          // 0..31
            int cseg = (t & 7) * 16;    // 0..112
            const float4* p = reinterpret_cast<const float4*>(
                feat + (blkRow0 + jb + jr) * DFEAT + cseg);
            float4 q0 = p[0], q1 = p[1], q2 = p[2], q3 = p[3];
            unsigned* dst = reinterpret_cast<unsigned*>(&jtile[jr][cseg]);
            dst[0] = pack2(q0.x, q0.y); dst[1] = pack2(q0.z, q0.w);
            dst[2] = pack2(q1.x, q1.y); dst[3] = pack2(q1.z, q1.w);
            dst[4] = pack2(q2.x, q2.y); dst[5] = pack2(q2.z, q2.w);
            dst[6] = pack2(q3.x, q3.y); dst[7] = pack2(q3.z, q3.w);
            float qv[16] = { q0.x, q0.y, q0.z, q0.w, q1.x, q1.y, q1.z, q1.w,
                             q2.x, q2.y, q2.z, q2.w, q3.x, q3.y, q3.z, q3.w };
            #pragma unroll
            for (int e = 0; e < 16; ++e) jtileT[cseg + e][jr] = (__bf16)qv[e];
        }
        __syncthreads();

        // S = X_i (16x128) * X_j^T (128x32): 2 N-halves x 4 K-steps = 8 WMMAs
        v8f s0 = {}, s1 = {};
        #pragma unroll
        for (int f = 0; f < 4; ++f) {
            v8u u0, u1;
            #pragma unroll
            for (int i = 0; i < 8; ++i) {
                int col = f * 32 + (i >> 2) * 16 + half * 8 + (i & 3) * 2;  // B layout: same K order as A
                u0[i] = *reinterpret_cast<const unsigned*>(&jtile[mrow][col]);
                u1[i] = *reinterpret_cast<const unsigned*>(&jtile[16 + mrow][col]);
            }
            v16bf b0 = __builtin_bit_cast(v16bf, u0);
            v16bf b1 = __builtin_bit_cast(v16bf, u1);
            s0 = __builtin_amdgcn_wmma_f32_16x16x32_bf16(false, aX[f], false, b0,
                                                         (short)0, s0, false, false);
            s1 = __builtin_amdgcn_wmma_f32_16x16x32_bf16(false, aX[f], false, b1,
                                                         (short)0, s1, false, false);
        }

        // Gaussian kernel weights in f32 on the C-layout, re-layout to A via LDS.
        float sj0 = sqj[jb + mrow];
        float sj1 = sqj[jb + 16 + mrow];
        #pragma unroll
        for (int v = 0; v < 8; ++v) {
            int M = v + half * 8;
            float d0 = fmaxf(si[v] + sj0 - 2.0f * s0[v], 0.0f);
            float d1 = fmaxf(si[v] + sj1 - 2.0f * s1[v], 0.0f);
            wtile[wave][M][mrow]      = (__bf16)__expf(-d0 * kscale);
            wtile[wave][M][16 + mrow] = (__bf16)__expf(-d1 * kscale);
        }
        // Same-wave LDS RAW: DS ops are in-order; compiler inserts dscnt waits.
        v8u uw;
        #pragma unroll
        for (int i = 0; i < 8; ++i) {
            int K = (i >> 2) * 16 + half * 8 + (i & 3) * 2;  // A layout
            uw[i] = *reinterpret_cast<const unsigned*>(&wtile[wave][mrow][K]);
        }
        v16bf aW = __builtin_bit_cast(v16bf, uw);

        // O += W_tile (16x32) * X_j (32x128): 8 N-tiles of 16 feature columns
        #pragma unroll
        for (int n = 0; n < 8; ++n) {
            v8u ub;
            #pragma unroll
            for (int i = 0; i < 8; ++i) {
                int dcol = n * 16 + mrow;
                int K    = (i >> 2) * 16 + half * 8 + (i & 3) * 2;
                ub[i] = *reinterpret_cast<const unsigned*>(&jtileT[dcol][K]);
            }
            v16bf bX = __builtin_bit_cast(v16bf, ub);
            accO[n] = __builtin_amdgcn_wmma_f32_16x16x32_bf16(false, aW, false, bX,
                                                              (short)0, accO[n], false, false);
        }
        __syncthreads();  // protect jtile/jtileT before next staging
    }

    // ---- Epilogue: out = accO / 1024, scattered per C/D layout
    const float inv = 1.0f / (float)BLK;
    #pragma unroll
    for (int n = 0; n < 8; ++n) {
        #pragma unroll
        for (int v = 0; v < 8; ++v) {
            int M = v + half * 8;
            long long row = blkRow0 + rLoc + M;
            out[row * DFEAT + n * 16 + mrow] = accO[n][v] * inv;
        }
    }
}

extern "C" void kernel_launch(void* const* d_in, const int* in_sizes, int n_in,
                              void* d_out, int out_size, void* d_ws, size_t ws_size,
                              hipStream_t stream) {
    (void)n_in; (void)d_ws; (void)ws_size;
    const float* feat = (const float*)d_in[0];
    float* out = (float*)d_out;

    const int total = in_sizes[0];       // N * 128
    const int N     = total / DFEAT;
    const int nb    = N / BLK;           // full 1024-row blocks

    // Reference leaves any tail rows as zeros (N=262144 has none, but be faithful).
    const long long covered = (long long)nb * BLK * DFEAT;
    if (covered < (long long)out_size) {
        hipMemsetAsync((char*)d_out + covered * sizeof(float), 0,
                       ((long long)out_size - covered) * sizeof(float), stream);
    }

    dim3 grid(nb * (BLK / ROWS_WG));     // 256 blocks x 8 WGs = 2048 workgroups
    block_attn_wmma<<<grid, 256, 0, stream>>>(feat, out);
}
